// CheferWeightedMHA_19739669692440
// MI455X (gfx1250) — matrix operand order repaired and checked by reading, not compiled, not run on hardware
//
#include <hip/hip_runtime.h>
#include <hip/hip_bf16.h>
#include <math.h>

// ---------------------------------------------------------------------------
// Problem constants (from the reference)
// ---------------------------------------------------------------------------
#define BB 4
#define SS 2048
#define HH 16
#define DK 64
#define DD (HH * DK)          // 1024
#define NEG_INF (-1e9f)
#define LOG2E 1.4426950408889634f

typedef __bf16 bf16_t;
typedef __attribute__((ext_vector_type(16))) __bf16 v16bf;
typedef __attribute__((ext_vector_type(2)))  __bf16 v2bf;
typedef __attribute__((ext_vector_type(8)))  float  v8f;
typedef __attribute__((ext_vector_type(2)))  float  v2f;

// K offset of VGPR v (pair base) inside a 16x32 bf16 A fragment.
// Lanes 0-15: V0..3 -> K=0..7, V4..7 -> K=16..23 ; lanes 16-31: +8.
__device__ __forceinline__ int a_koff(int v, int half) {
  return ((v >> 2) * 16) + (half * 8) + ((v & 3) * 2);
}
// K offset of VGPR v inside a 32x16 bf16 B fragment (same order as A).
// Lanes 0-15: V0..3 -> K=0..7, V4..7 -> K=16..23 ; lanes 16-31: +8.
__device__ __forceinline__ int b_koff(int v, int half) {
  return ((v >> 2) * 16) + (half * 8) + ((v & 3) * 2);
}

__device__ __forceinline__ v8f wmma_bf16(v16bf a, v16bf b, v8f c) {
  return __builtin_amdgcn_wmma_f32_16x16x32_bf16(
      /*neg_a=*/false, a, /*neg_b=*/false, b,
      /*c_mod=*/(short)0, c, /*reuse_a=*/false, /*reuse_b=*/false);
}

// ---------------------------------------------------------------------------
// Kernel 1: v = value @ V_w^T + V_b, written TRANSPOSED as vT[b][h][d][s] bf16
//   M = B*S = 8192, N = D = 1024, K = D = 1024
//   One wave per (16-row, 64-col) tile; 4 WMMA accumulators per wave.
//   All 4 B fragments are materialized before the 4 WMMAs so the matrix ops
//   issue back-to-back with no WAR hazard nops on their source registers.
// ---------------------------------------------------------------------------
__global__ __launch_bounds__(256) void gemm_vproj_kernel(
    const float* __restrict__ value, const float* __restrict__ Vw,
    const float* __restrict__ Vb, bf16_t* __restrict__ vt)
{
  const int lane = threadIdx.x & 31;
  const int wave = threadIdx.x >> 5;
  const int gw   = blockIdx.x * 8 + wave;      // 8192 waves total
  const int m0   = (gw >> 4) * 16;             // row tile base (0..8176)
  const int head = gw & 15;                    // 64-col group == head index
  const int n0   = head * 64;
  const int half = lane >> 4;
  const int ln   = lane & 15;

  v8f c[4];
  #pragma unroll
  for (int t = 0; t < 4; ++t) c[t] = (v8f)0.f;

  const float* arow = value + (size_t)(m0 + ln) * DD;

  for (int k0 = 0; k0 < DD; k0 += 32) {
    v16bf a;
    #pragma unroll
    for (int v = 0; v < 8; ++v) {
      const int K = a_koff(v, half);
      v2f t = *(const v2f*)(arow + k0 + K);
      a[2 * v]     = (bf16_t)t[0];
      a[2 * v + 1] = (bf16_t)t[1];
    }
    v16bf bm[4];
    #pragma unroll
    for (int t = 0; t < 4; ++t) {
      const float* brow = Vw + (size_t)(n0 + t * 16 + ln) * DD + k0;
      #pragma unroll
      for (int v = 0; v < 8; ++v) {
        v2f u = *(const v2f*)(brow + b_koff(v, half));
        bm[t][2 * v]     = (bf16_t)u[0];
        bm[t][2 * v + 1] = (bf16_t)u[1];
      }
    }
    #pragma unroll
    for (int t = 0; t < 4; ++t) c[t] = wmma_bf16(a, bm[t], c[t]);
  }

  // Epilogue: add bias, store transposed bf16: vt[((b*H + head)*DK + d)*S + s]
  const int b = m0 / SS;                       // whole tile stays in one batch
  #pragma unroll
  for (int t = 0; t < 4; ++t) {
    const int d = t * 16 + ln;
    const float bias = Vb[n0 + d];
    #pragma unroll
    for (int r = 0; r < 8; ++r) {
      const int m = m0 + r + half * 8;
      const int s = m & (SS - 1);
      vt[(((size_t)b * HH + head) * DK + d) * SS + s] = (bf16_t)(c[t][r] + bias);
    }
  }
}

// ---------------------------------------------------------------------------
// Kernel 2: fused flash softmax + PV:  x[b,s,h,d] = softmax(mask?w:-1e9) @ v
//   One wave owns 16 query rows and the full d_k=64 (4 WMMA tiles),
//   streaming K in 32-wide tiles with online max/sum rescaling.
//   weight is read exactly ONCE (the dominant ~1.07 GB of HBM traffic).
//   The accumulator rescale is skipped (wave-uniform branch) whenever the
//   running max did not change, which is the common case.
// ---------------------------------------------------------------------------
__global__ __launch_bounds__(128) void attn_softmax_pv_kernel(
    const float* __restrict__ weight, const unsigned char* __restrict__ mask,
    const bf16_t* __restrict__ vt, bf16_t* __restrict__ xbf)
{
  const int lane   = threadIdx.x & 31;
  const int wave   = threadIdx.x >> 5;
  const int qblks  = SS / 64;
  const int qblk   = blockIdx.x % qblks;
  const int bh     = blockIdx.x / qblks;
  const int b      = bh / HH;
  const int h      = bh % HH;
  const int q0     = qblk * 64 + wave * 16;
  const int half   = lane >> 4;
  const int ln     = lane & 15;
  const int qrow   = q0 + ln;

  const float*         wrow  = weight + ((size_t)bh * SS + qrow) * SS;
  const unsigned char* mrow  = mask   + ((size_t)b  * SS + qrow) * SS;
  const bf16_t*        vbase = vt     + (size_t)bh * DK * SS;

  v8f c[4];
  #pragma unroll
  for (int t = 0; t < 4; ++t) c[t] = (v8f)0.f;

  float run_max = -INFINITY;
  float run_sum = 0.f;

  for (int k0 = 0; k0 < SS; k0 += 32) {
    // Prefetch the weight stream ~4 tiles (512B/row) ahead.
    {
      const int kpf = (k0 + 128 < SS) ? (k0 + 128) : (SS - 32);
      __builtin_prefetch(wrow + kpf, 0, 0);
    }

    // ---- load 16 scores in A-fragment layout, apply mask ----
    float w[16];
    #pragma unroll
    for (int v = 0; v < 8; ++v) {
      const int K = a_koff(v, half);
      v2f t = *(const v2f*)(wrow + k0 + K);
      const unsigned short mp = *(const unsigned short*)(mrow + k0 + K);
      w[2 * v]     = (mp & 0x00ff) ? t[0] : NEG_INF;
      w[2 * v + 1] = (mp & 0xff00) ? t[1] : NEG_INF;
    }
    // ---- tile row max (16 in-lane + cross-half shuffle) ----
    float tmax = w[0];
    #pragma unroll
    for (int i = 1; i < 16; ++i) tmax = fmaxf(tmax, w[i]);
    tmax = fmaxf(tmax, __shfl_xor(tmax, 16));

    // Wave-uniform: did ANY row's running max change?
    if (__builtin_amdgcn_ballot_w32(tmax > run_max) != 0u) {
      const float nmax  = fmaxf(run_max, tmax);
      const float alpha = __builtin_amdgcn_exp2f((run_max - nmax) * LOG2E);
      run_max = nmax;
      run_sum *= alpha;
      float ar[8];
      #pragma unroll
      for (int r = 0; r < 8; ++r) ar[r] = __shfl(alpha, r + half * 8);
      #pragma unroll
      for (int t = 0; t < 4; ++t)
        #pragma unroll
        for (int r = 0; r < 8; ++r) c[t][r] *= ar[r];
    }

    // ---- p = exp(w - run_max), row partial sum, pack to bf16 A fragment ----
    v16bf a;
    float psum = 0.f;
    #pragma unroll
    for (int i = 0; i < 16; ++i) {
      const float p = __builtin_amdgcn_exp2f((w[i] - run_max) * LOG2E);
      psum += p;
      a[i] = (bf16_t)p;
    }
    psum += __shfl_xor(psum, 16);
    run_sum += psum;

    // ---- gather all 4 B fragments (vT rows contiguous in s), then WMMAs ----
    v16bf bm[4];
    #pragma unroll
    for (int t = 0; t < 4; ++t) {
      const bf16_t* vr = vbase + (size_t)(t * 16 + ln) * SS + k0;
      #pragma unroll
      for (int v = 0; v < 8; ++v) {
        v2bf u = *(const v2bf*)(vr + b_koff(v, half));
        bm[t][2 * v]     = u[0];
        bm[t][2 * v + 1] = u[1];
      }
    }
    #pragma unroll
    for (int t = 0; t < 4; ++t) c[t] = wmma_bf16(a, bm[t], c[t]);
  }

  // ---- finalize: divide by row sum, store x[b,q,h,d] as bf16 ----
  const float inv = 1.f / run_sum;
  float sr[8];
  #pragma unroll
  for (int r = 0; r < 8; ++r) sr[r] = __shfl(inv, r + half * 8);

  #pragma unroll
  for (int t = 0; t < 4; ++t) {
    const int d = t * 16 + ln;
    #pragma unroll
    for (int r = 0; r < 8; ++r) {
      const int q = q0 + r + half * 8;
      xbf[(((size_t)b * SS + q) * HH + h) * DK + d] = (bf16_t)(c[t][r] * sr[r]);
    }
  }
}

// ---------------------------------------------------------------------------
// Kernel 3: out = x @ O_w^T + O_b  (x in bf16, output fp32)
// ---------------------------------------------------------------------------
__global__ __launch_bounds__(256) void gemm_oproj_kernel(
    const bf16_t* __restrict__ xbf, const float* __restrict__ Ow,
    const float* __restrict__ Ob, float* __restrict__ out)
{
  const int lane = threadIdx.x & 31;
  const int wave = threadIdx.x >> 5;
  const int gw   = blockIdx.x * 8 + wave;
  const int m0   = (gw >> 4) * 16;
  const int n0   = (gw & 15) * 64;
  const int half = lane >> 4;
  const int ln   = lane & 15;

  v8f c[4];
  #pragma unroll
  for (int t = 0; t < 4; ++t) c[t] = (v8f)0.f;

  const bf16_t* arow = xbf + (size_t)(m0 + ln) * DD;

  for (int k0 = 0; k0 < DD; k0 += 32) {
    v16bf a;
    #pragma unroll
    for (int v = 0; v < 8; ++v) {
      v2bf t = *(const v2bf*)(arow + k0 + a_koff(v, half));
      a[2 * v]     = t[0];
      a[2 * v + 1] = t[1];
    }
    v16bf bm[4];
    #pragma unroll
    for (int t = 0; t < 4; ++t) {
      const float* brow = Ow + (size_t)(n0 + t * 16 + ln) * DD + k0;
      #pragma unroll
      for (int v = 0; v < 8; ++v) {
        v2f u = *(const v2f*)(brow + b_koff(v, half));
        bm[t][2 * v]     = (bf16_t)u[0];
        bm[t][2 * v + 1] = (bf16_t)u[1];
      }
    }
    #pragma unroll
    for (int t = 0; t < 4; ++t) c[t] = wmma_bf16(a, bm[t], c[t]);
  }

  #pragma unroll
  for (int t = 0; t < 4; ++t) {
    const int n = n0 + t * 16 + ln;
    const float bias = Ob[n];
    #pragma unroll
    for (int r = 0; r < 8; ++r) {
      const int m = m0 + r + half * 8;
      out[(size_t)m * DD + n] = c[t][r] + bias;
    }
  }
}

// ---------------------------------------------------------------------------
// Host launcher
// Inputs (setup_inputs order): 0=query 1=key 2=value 3=weight 4=mask
//                              5=V_w 6=V_b 7=O_w 8=O_b
// Workspace: [0,16MB) vT bf16 [B,H,DK,S] ; [16MB,32MB) x bf16 [B,S,H,DK]
// ---------------------------------------------------------------------------
extern "C" void kernel_launch(void* const* d_in, const int* in_sizes, int n_in,
                              void* d_out, int out_size, void* d_ws, size_t ws_size,
                              hipStream_t stream) {
  (void)in_sizes; (void)n_in; (void)out_size; (void)ws_size;

  const float*         value  = (const float*)d_in[2];
  const float*         weight = (const float*)d_in[3];
  const unsigned char* mask   = (const unsigned char*)d_in[4];
  const float*         Vw     = (const float*)d_in[5];
  const float*         Vb     = (const float*)d_in[6];
  const float*         Ow     = (const float*)d_in[7];
  const float*         Ob     = (const float*)d_in[8];
  float*               outp   = (float*)d_out;

  bf16_t* vt  = (bf16_t*)d_ws;                                     // 16 MB
  bf16_t* xbf = (bf16_t*)((char*)d_ws + (size_t)16 * 1024 * 1024); // 16 MB

  // GEMM tiling: (M/16)*(N/64) = 512*16 = 8192 waves, 8 waves/block.
  const int gemm_blocks = 8192 / 8;
  gemm_vproj_kernel<<<gemm_blocks, 256, 0, stream>>>(value, Vw, Vb, vt);

  // Attention: B*H*(S/64) workgroups of 4 waves (each wave: 16 q rows, full d_k).
  const int attn_blocks = BB * HH * (SS / 64);
  attn_softmax_pv_kernel<<<attn_blocks, 128, 0, stream>>>(weight, mask, vt, xbf);

  gemm_oproj_kernel<<<gemm_blocks, 256, 0, stream>>>(xbf, Ow, Ob, outp);
}
